// CG_84817014162012
// MI455X (gfx1250) — compile-verified, matrix operand order checked by reading, not run on hardware
//
#include <hip/hip_runtime.h>
#include <math.h>

// ---------------------------------------------------------------------------
// Problem constants (from the reference)
// ---------------------------------------------------------------------------
#define NP_  20000
#define NA_  30000
#define NF_  8000
#define NV_  4000
#define HH   512
#define PRJ  256
#define NMAXN 30000

typedef __attribute__((ext_vector_type(16))) _Float16 v16h;
typedef __attribute__((ext_vector_type(2)))  _Float16 v2h;
typedef __attribute__((ext_vector_type(8)))  float    v8f;

// ---------------------------------------------------------------------------
// WMMA GEMM:  C[M,N] = act(A[M,K] @ B[K,N] + bias)
//   A,B,C row-major f32; f32->f16 staged via LDS; f32 accumulate in WMMA.
//   act: 0=none, 1=PReLU(*alpha), 2=ReLU, 3=tanh
//   Requires: K % 32 == 0, N % 128 == 0 (true for all call sites here).
// ---------------------------------------------------------------------------
#define BM 128
#define BN 128
#define BK 32
#define LDH 48   // halves per LDS row: 32 data + 16 pad -> 96B rows, 32B aligned

__global__ __launch_bounds__(256)
void hg_gemm_wmma(const float* __restrict__ A, const float* __restrict__ B,
                  const float* __restrict__ bias, const float* __restrict__ alpha,
                  float* __restrict__ C, int M, int N, int K, int act)
{
    __shared__ __align__(32) _Float16 As[BM * LDH];
    __shared__ __align__(32) _Float16 Bs[BN * LDH];   // Bs[n][k] = B[k][n]

    const int tid  = threadIdx.x;
    const int lane = tid & 31;
    const int wave = tid >> 5;
    const int bm = blockIdx.y * BM;
    const int bn = blockIdx.x * BN;

    const int wm = (wave >> 1) * 32;    // wave sub-tile: 32 rows x 64 cols
    const int wn = (wave & 1) * 64;

    v8f acc[2][4];
#pragma unroll
    for (int i = 0; i < 2; ++i)
#pragma unroll
        for (int j = 0; j < 4; ++j) acc[i][j] = (v8f)0.0f;

    const int half = lane >> 4;   // K-half select for fragments
    const int l16  = lane & 15;

    for (int k0 = 0; k0 < K; k0 += BK) {
        // Prefetch next A tile rows (speculative; lowers to global_prefetch_b8)
        if (k0 + BK < K) {
            int pr = bm + (tid >> 1);
            if (pr < M) __builtin_prefetch(&A[(size_t)pr * K + k0 + BK], 0, 1);
        }
        // Stage A tile 128x32 (f32 -> f16): float2 loads, packed b32 LDS stores
        for (int t = tid; t < (BM * BK) / 2; t += 256) {
            int r  = t >> 4;            // row 0..127
            int c2 = (t & 15) * 2;      // even k within tile
            int gr = bm + r;
            float v0 = 0.f, v1 = 0.f;
            if (gr < M) {
                float2 f2 = *(const float2*)&A[(size_t)gr * K + (k0 + c2)];
                v0 = f2.x; v1 = f2.y;
            }
            v2h p; p.x = (_Float16)v0; p.y = (_Float16)v1;
            *(v2h*)&As[r * LDH + c2] = p;
        }
        // Stage B tile 32x128 transposed: pack two k-adjacent column elems
        for (int t = tid; t < (BK * BN) / 2; t += 256) {
            int n  = t & 127;           // column within tile
            int k2 = (t >> 7) * 2;      // even k within tile
            float v0 = B[(size_t)(k0 + k2)     * N + (bn + n)];
            float v1 = B[(size_t)(k0 + k2 + 1) * N + (bn + n)];
            v2h p; p.x = (_Float16)v0; p.y = (_Float16)v1;
            *(v2h*)&Bs[n * LDH + k2] = p;
        }
        __syncthreads();

        v16h af[2], bf[4];
#pragma unroll
        for (int i = 0; i < 2; ++i)
            af[i] = *(const v16h*)&As[(wm + i * 16 + l16) * LDH + half * 16];
#pragma unroll
        for (int j = 0; j < 4; ++j)
            bf[j] = *(const v16h*)&Bs[(wn + j * 16 + l16) * LDH + half * 16];
#pragma unroll
        for (int i = 0; i < 2; ++i)
#pragma unroll
            for (int j = 0; j < 4; ++j)
                acc[i][j] = __builtin_amdgcn_wmma_f32_16x16x32_f16(
                    false, af[i], false, bf[j], (short)0, acc[i][j], false, false);
        __syncthreads();
    }

    // Epilogue: C layout per lane = (row = v + 8*(lane>>4), col = lane&15)
    const int hi = lane >> 4;
    const int n16 = lane & 15;
#pragma unroll
    for (int i = 0; i < 2; ++i) {
#pragma unroll
        for (int j = 0; j < 4; ++j) {
            int col = bn + wn + j * 16 + n16;
            float bz = bias ? bias[col] : 0.0f;
#pragma unroll
            for (int r = 0; r < 8; ++r) {
                int row = bm + wm + i * 16 + hi * 8 + r;
                if (row < M) {
                    float v = acc[i][j][r] + bz;
                    if (act == 1)      { float a = *alpha; v = v >= 0.f ? v : a * v; }
                    else if (act == 2) { v = fmaxf(v, 0.f); }
                    else if (act == 3) { v = tanhf(v); }
                    C[(size_t)row * N + col] = v;
                }
            }
        }
    }
}

// ---------------------------------------------------------------------------
// Column mean / biased variance over M rows (one block per column)
// ---------------------------------------------------------------------------
__global__ __launch_bounds__(256)
void hg_colstats(const float* __restrict__ X, int M, int N,
                 float* __restrict__ mean, float* __restrict__ var)
{
    __shared__ float s0[256], s1[256];
    int col = blockIdx.x;
    float s = 0.f, ss = 0.f;
    for (int r = threadIdx.x; r < M; r += blockDim.x) {
        float v = X[(size_t)r * N + col];
        s += v; ss += v * v;
    }
    s0[threadIdx.x] = s; s1[threadIdx.x] = ss;
    __syncthreads();
    for (int o = 128; o > 0; o >>= 1) {
        if (threadIdx.x < o) {
            s0[threadIdx.x] += s0[threadIdx.x + o];
            s1[threadIdx.x] += s1[threadIdx.x + o];
        }
        __syncthreads();
    }
    if (threadIdx.x == 0) {
        float m = s0[0] / (float)M;
        mean[col] = m;
        var[col]  = s1[0] / (float)M - m * m;
    }
}

// BN apply + optional activation. act: 0 none, 1 prelu, 2 relu
__global__ void hg_bn_act(float* __restrict__ X, long long total, int N,
                          const float* __restrict__ mean, const float* __restrict__ var,
                          const float* __restrict__ g, const float* __restrict__ b,
                          const float* __restrict__ alpha, int act)
{
    long long i = (long long)blockIdx.x * blockDim.x + threadIdx.x;
    if (i >= total) return;
    int col = (int)(i % N);
    float v = X[i];
    v = g[col] * (v - mean[col]) * rsqrtf(var[col] + 1e-5f) + b[col];
    if (act == 1)      { float a = *alpha; v = v >= 0.f ? v : a * v; }
    else if (act == 2) { v = fmaxf(v, 0.f); }
    X[i] = v;
}

__global__ void hg_fill(float* __restrict__ p, long long n, float v)
{
    long long i = (long long)blockIdx.x * blockDim.x + threadIdx.x;
    if (i < n) p[i] = v;
}

// out[dst] += X[src] (512-wide), cnt[dst] += 1   (one block per edge)
__global__ void hg_seg_scatter(const float* __restrict__ X, const int* __restrict__ src,
                               const int* __restrict__ dst, int E, int Hd,
                               float* __restrict__ out, float* __restrict__ cnt)
{
    int e = blockIdx.x;
    if (e >= E) return;
    int s = src[e], d = dst[e];
    const float* xr = X + (size_t)s * Hd;
    float* orow = out + (size_t)d * Hd;
    for (int j = threadIdx.x; j < Hd; j += blockDim.x) atomicAdd(&orow[j], xr[j]);
    if (threadIdx.x == 0) atomicAdd(&cnt[d], 1.0f);
}

__global__ void hg_div_cnt(float* __restrict__ X, const float* __restrict__ cnt,
                           long long total, int N)
{
    long long i = (long long)blockIdx.x * blockDim.x + threadIdx.x;
    if (i >= total) return;
    int row = (int)(i / N);
    X[i] /= fmaxf(cnt[row], 1.0f);
}

// out[row] = dot(X[row,:], w)  (one wave per row)
__global__ void hg_rowdot(const float* __restrict__ X, const float* __restrict__ w,
                          float* __restrict__ out, int M, int N)
{
    int row = blockIdx.x * (blockDim.x >> 5) + (threadIdx.x >> 5);
    int lane = threadIdx.x & 31;
    if (row >= M) return;
    float s = 0.f;
    for (int j = lane; j < N; j += 32) s += X[(size_t)row * N + j] * w[j];
    for (int o = 16; o > 0; o >>= 1) s += __shfl_down(s, o, 32);
    if (lane == 0) out[row] = s;
}

__device__ inline void hg_atomicMaxF(float* addr, float val)
{
    int* ai = (int*)addr;
    int old = *ai;
    while (val > __int_as_float(old)) {
        int assumed = old;
        old = atomicCAS(ai, assumed, __float_as_int(val));
        if (old == assumed) break;
    }
}

__global__ void hg_edge_logit(const float* __restrict__ ss, const float* __restrict__ sd,
                              const int* __restrict__ e0, const int* __restrict__ e1,
                              int E, float* __restrict__ ea, float* __restrict__ nm)
{
    int e = blockIdx.x * blockDim.x + threadIdx.x;
    if (e >= E) return;
    float a = ss[e0[e]] + sd[e1[e]];
    a = a >= 0.f ? a : 0.01f * a;            // leaky_relu, slope 0.01
    ea[e] = a;
    hg_atomicMaxF(&nm[e1[e]], a);
}

__global__ void hg_fix_max(float* __restrict__ nm, int n)
{
    int i = blockIdx.x * blockDim.x + threadIdx.x;
    if (i >= n) return;
    float v = nm[i];
    if (!__builtin_isfinite(v)) nm[i] = 0.f;
}

__global__ void hg_edge_exp(const float* __restrict__ ea, const float* __restrict__ nm,
                            const int* __restrict__ e1, int E,
                            float* __restrict__ ew, float* __restrict__ ns)
{
    int e = blockIdx.x * blockDim.x + threadIdx.x;
    if (e >= E) return;
    float w = expf(ea[e] - nm[e1[e]]);
    ew[e] = w;
    atomicAdd(&ns[e1[e]], w);
}

__global__ void hg_gat_scatter(const float* __restrict__ X, const int* __restrict__ src,
                               const int* __restrict__ dst, const float* __restrict__ ew,
                               const float* __restrict__ ns, int E, int Hd,
                               float* __restrict__ out)
{
    int e = blockIdx.x;
    if (e >= E) return;
    int s = src[e], d = dst[e];
    float w = ew[e] / (ns[d] + 1e-16f);
    const float* xr = X + (size_t)s * Hd;
    float* orow = out + (size_t)d * Hd;
    for (int j = threadIdx.x; j < Hd; j += blockDim.x) atomicAdd(&orow[j], xr[j] * w);
}

__global__ void hg_add_bias(float* __restrict__ X, const float* __restrict__ b,
                            long long total, int N)
{
    long long i = (long long)blockIdx.x * blockDim.x + threadIdx.x;
    if (i >= total) return;
    X[i] += b[(int)(i % N)];
}

// beta = softmax_m( sp[m,:] . att )   (single block, one wave per m)
__global__ void hg_agg_beta(const float* __restrict__ sp, const float* __restrict__ att,
                            float* __restrict__ beta)
{
    __shared__ float sh[3];
    int m = threadIdx.x >> 5;
    int lane = threadIdx.x & 31;
    if (m < 3) {
        float s = 0.f;
        for (int j = lane; j < HH; j += 32) s += sp[m * HH + j] * att[j];
        for (int o = 16; o > 0; o >>= 1) s += __shfl_down(s, o, 32);
        if (lane == 0) sh[m] = s;
    }
    __syncthreads();
    if (threadIdx.x == 0) {
        float mx = fmaxf(sh[0], fmaxf(sh[1], sh[2]));
        float e0 = expf(sh[0] - mx), e1 = expf(sh[1] - mx), e2 = expf(sh[2] - mx);
        float s = e0 + e1 + e2;
        beta[0] = e0 / s; beta[1] = e1 / s; beta[2] = e2 / s;
    }
}

__global__ void hg_wsum3(const float* __restrict__ beta,
                         const float* __restrict__ a, const float* __restrict__ b,
                         const float* __restrict__ c, float* __restrict__ o, long long n)
{
    long long i = (long long)blockIdx.x * blockDim.x + threadIdx.x;
    if (i >= n) return;
    o[i] = beta[0] * a[i] + beta[1] * b[i] + beta[2] * c[i];
}

__global__ void hg_add2(const float* __restrict__ a, const float* __restrict__ b,
                        float* __restrict__ o, long long n)
{
    long long i = (long long)blockIdx.x * blockDim.x + threadIdx.x;
    if (i >= n) return;
    o[i] = a[i] + b[i];
}

// mean(1 - cos(x,y)) accumulated into *acc  (one wave per row)
__global__ void hg_sce(const float* __restrict__ X, const float* __restrict__ Y,
                       int M, int N, float* __restrict__ acc)
{
    int row = blockIdx.x * (blockDim.x >> 5) + (threadIdx.x >> 5);
    int lane = threadIdx.x & 31;
    if (row >= M) return;
    const float* x = X + (size_t)row * N;
    const float* y = Y + (size_t)row * N;
    float xx = 0.f, yy = 0.f, xy = 0.f;
    for (int j = lane; j < N; j += 32) {
        float a = x[j], b = y[j];
        xx += a * a; yy += b * b; xy += a * b;
    }
    for (int o = 16; o > 0; o >>= 1) {
        xx += __shfl_down(xx, o, 32);
        yy += __shfl_down(yy, o, 32);
        xy += __shfl_down(xy, o, 32);
    }
    if (lane == 0) {
        float inv = rsqrtf(fmaxf(xx, 1e-24f)) * rsqrtf(fmaxf(yy, 1e-24f));
        atomicAdd(acc, (1.0f - xy * inv) / (float)M);
    }
}

// out = L1 + L5 + 0.5*(L2+L3+L4);  L = [l1,l2,l3,l4,l5,...]
__global__ void hg_combine(const float* __restrict__ L, float* __restrict__ out)
{
    if (blockIdx.x == 0 && threadIdx.x == 0)
        out[0] = L[0] + L[4] + 0.5f * (L[1] + L[2] + L[3]);
}

// ---------------------------------------------------------------------------
// Host orchestration
// ---------------------------------------------------------------------------
extern "C" void kernel_launch(void* const* d_in, const int* in_sizes, int n_in,
                              void* d_out, int out_size, void* d_ws, size_t ws_size,
                              hipStream_t stream)
{
    (void)n_in; (void)out_size; (void)ws_size;
    const int NP = NP_, NA = NA_, NF = NF_, NV = NV_, H = HH, PR = PRJ;

    const float* xP = (const float*)d_in[0];
    const float* xA = (const float*)d_in[1];
    const float* xF = (const float*)d_in[2];
    const float* xV = (const float*)d_in[3];
    const int* eAP = (const int*)d_in[4]; const int E_AP = in_sizes[4] / 2;
    const int* eFP = (const int*)d_in[5]; const int E_FP = in_sizes[5] / 2;
    const int* eVP = (const int*)d_in[6]; const int E_VP = in_sizes[6] / 2;
    const int* ePA = (const int*)d_in[7]; const int E_PA = in_sizes[7] / 2;
    const int* ePF = (const int*)d_in[8]; const int E_PF = in_sizes[8] / 2;
    const int* ePV = (const int*)d_in[9]; const int E_PV = in_sizes[9] / 2;

    // Param leaves in JAX pytree (sorted-key) order, starting at d_in[10]:
    //  10..12  agg1: W, att, b         13..15  agg2: W, att, b
    //  16..27  fc: A{W,a,b} F{W,a,b} P{W,a,b} V{W,a,b}
    //  28..45  gat[0..2]: a, att_d, att_s, bias, bn.b, bn.g
    //  46..69  proj1..proj4: bn.b, bn.g, l1.W, l1.b, l2.W, l2.b
    //  70..99  sage[0..2]:  a, a1, bn.b, bn.g, bn1.b, bn1.g, lin.W, lin.b, lin2.W, lin2.b
    // 100..129 sage1[0..2]: same layout
    auto PP = [&](int i) { return (const float*)d_in[i]; };

    float* ws = (float*)d_ws;
    size_t off = 0;
    auto alloc = [&](size_t n) { float* p = ws + off; off += n; return p; };

    float* hP   = alloc((size_t)NP * H);
    float* hA   = alloc((size_t)NA * H);
    float* hF   = alloc((size_t)NF * H);
    float* hV   = alloc((size_t)NV * H);
    float* t0   = alloc((size_t)NMAXN * H);
    float* t1   = alloc((size_t)NMAXN * H);
    float* t2   = alloc((size_t)NMAXN * H);
    float* h2b[3]; for (int i = 0; i < 3; ++i) h2b[i] = alloc((size_t)NP * H);
    float* seb[3]; for (int i = 0; i < 3; ++i) seb[i] = alloc((size_t)NP * H);
    float* emb1 = alloc((size_t)NP * H);
    float* emb2 = alloc((size_t)NP * H);
    float* pt   = alloc((size_t)NP * PR);
    float* po   = alloc((size_t)NP * H);
    float* tgt  = alloc((size_t)NP * H);
    float* cnt  = alloc(NMAXN);
    float* sa   = alloc(NMAXN);
    float* sd   = alloc(NP);
    float* nm   = alloc(NP);
    float* ns   = alloc(NP);
    float* mean = alloc(H);
    float* var  = alloc(H);
    float* sp   = alloc(3 * H);
    float* beta = alloc(4);
    float* ea   = alloc(150000);
    float* ew   = alloc(150000);
    float* Lacc = alloc(8);

    auto gemm = [&](const float* A, const float* B, const float* bias, const float* alpha,
                    float* C, int M, int N, int K, int act) {
        dim3 g((unsigned)((N + BN - 1) / BN), (unsigned)((M + BM - 1) / BM));
        hg_gemm_wmma<<<g, 256, 0, stream>>>(A, B, bias, alpha, C, M, N, K, act);
    };
    auto stats = [&](const float* X, int M, int N, float* mn, float* vr) {
        hg_colstats<<<N, 256, 0, stream>>>(X, M, N, mn, vr);
    };
    auto bnact = [&](float* X, int M, int N, const float* g, const float* b,
                     const float* alpha, int act) {
        long long tot = (long long)M * N;
        hg_bn_act<<<(unsigned)((tot + 255) / 256), 256, 0, stream>>>(
            X, tot, N, mean, var, g, b, alpha, act);
    };
    auto fill = [&](float* p, long long n, float v) {
        hg_fill<<<(unsigned)((n + 255) / 256), 256, 0, stream>>>(p, n, v);
    };

    // ---- fc layers: h[nt] = prelu(x @ W + b, a) ----
    gemm(xP, PP(22), PP(24), PP(23), hP, NP, H, 1024, 1);
    gemm(xA, PP(16), PP(18), PP(17), hA, NA, H, 512, 1);
    gemm(xF, PP(19), PP(21), PP(20), hF, NF, H, 256, 1);
    gemm(xV, PP(25), PP(27), PP(26), hV, NV, H, 256, 1);

    // ---- one SAGE module: out = prelu(bn1(lin2(mean_agg(prelu(bn(lin(X))))))) ----
    auto sage_pass = [&](const float* X, int Nsrc, const int* esrc, const int* edst, int E,
                         int Ndst, int pb, float* outbuf) {
        gemm(X, PP(pb + 6), PP(pb + 7), nullptr, t0, Nsrc, H, H, 0);
        stats(t0, Nsrc, H, mean, var);
        bnact(t0, Nsrc, H, PP(pb + 3), PP(pb + 2), PP(pb + 0), 1);
        fill(t1, (long long)Ndst * H, 0.f);
        fill(cnt, Ndst, 0.f);
        hg_seg_scatter<<<E, 256, 0, stream>>>(t0, esrc, edst, E, H, t1, cnt);
        long long tot = (long long)Ndst * H;
        hg_div_cnt<<<(unsigned)((tot + 255) / 256), 256, 0, stream>>>(t1, cnt, tot, H);
        gemm(t1, PP(pb + 8), PP(pb + 9), nullptr, outbuf, Ndst, H, H, 0);
        stats(outbuf, Ndst, H, mean, var);
        bnact(outbuf, Ndst, H, PP(pb + 5), PP(pb + 4), PP(pb + 1), 1);
    };

    // ---- metapath SAGE branches: P -> X -> P ----
    {
        const int*  evs[3] = { ePA, ePF, ePV };
        const int   Es[3]  = { E_PA, E_PF, E_PV };
        const int   Nds[3] = { NA, NF, NV };
        for (int i = 0; i < 3; ++i) {
            const int* e0 = evs[i];              // P-side indices
            const int* e1 = evs[i] + Es[i];      // dst-type indices
            sage_pass(hP, NP, e0, e1, Es[i], Nds[i], 70 + 10 * i, t2);
            sage_pass(t2, Nds[i], e1, e0, Es[i], NP, 100 + 10 * i, h2b[i]);
        }
    }

    // ---- semantic aggregation ----
    auto agg = [&](float* const* views, int wb, float* outp) {
        for (int m = 0; m < 3; ++m) {
            gemm(views[m], PP(wb + 0), PP(wb + 2), nullptr, t0, NP, H, H, 3 /*tanh*/);
            stats(t0, NP, H, sp + m * H, var);     // column mean -> sp row m
        }
        hg_agg_beta<<<1, 128, 0, stream>>>(sp, PP(wb + 1), beta);
        long long tot = (long long)NP * H;
        hg_wsum3<<<(unsigned)((tot + 255) / 256), 256, 0, stream>>>(
            beta, views[0], views[1], views[2], outp, tot);
    };
    agg(h2b, 10, emb1);

    // ---- schema GAT branches: X -> P ----
    auto gat_pass = [&](const float* hS, int Nsrc, const int* e0, const int* e1, int E,
                        int gb, float* outp) {
        hg_rowdot<<<(Nsrc + 7) / 8, 256, 0, stream>>>(hS, PP(gb + 2), sa, Nsrc, H);
        hg_rowdot<<<(NP + 7) / 8, 256, 0, stream>>>(hP, PP(gb + 1), sd, NP, H);
        fill(nm, NP, -__builtin_inff());
        hg_edge_logit<<<(E + 255) / 256, 256, 0, stream>>>(sa, sd, e0, e1, E, ea, nm);
        hg_fix_max<<<(NP + 255) / 256, 256, 0, stream>>>(nm, NP);
        fill(ns, NP, 0.f);
        hg_edge_exp<<<(E + 255) / 256, 256, 0, stream>>>(ea, nm, e1, E, ew, ns);
        fill(outp, (long long)NP * H, 0.f);
        hg_gat_scatter<<<E, 256, 0, stream>>>(hS, e0, e1, ew, ns, E, H, outp);
        long long tot = (long long)NP * H;
        hg_add_bias<<<(unsigned)((tot + 255) / 256), 256, 0, stream>>>(outp, PP(gb + 3), tot, H);
        stats(outp, NP, H, mean, var);
        bnact(outp, NP, H, PP(gb + 5), PP(gb + 4), PP(gb + 0), 1);
    };
    gat_pass(hA, NA, eAP, eAP + E_AP, E_AP, 28, seb[0]);
    gat_pass(hF, NF, eFP, eFP + E_FP, E_FP, 34, seb[1]);
    gat_pass(hV, NV, eVP, eVP + E_VP, E_VP, 40, seb[2]);
    agg(seb, 13, emb2);

    // ---- projection heads + SCE losses ----
    auto proj = [&](const float* X, int pb, float* outp) {
        gemm(X, PP(pb + 2), PP(pb + 3), nullptr, pt, NP, PR, H, 0);
        stats(pt, NP, PR, mean, var);
        bnact(pt, NP, PR, PP(pb + 1), PP(pb + 0), nullptr, 2 /*relu*/);
        gemm(pt, PP(pb + 4), PP(pb + 5), nullptr, outp, NP, H, PR, 0);
    };
    auto sce = [&](const float* X, const float* Y, float* acc) {
        hg_sce<<<(NP + 7) / 8, 256, 0, stream>>>(X, Y, NP, H, acc);
    };
    auto add2 = [&](const float* a, const float* b, float* o) {
        long long tot = (long long)NP * H;
        hg_add2<<<(unsigned)((tot + 255) / 256), 256, 0, stream>>>(a, b, o, tot);
    };

    fill(Lacc, 8, 0.f);
    proj(emb2, 46, po); sce(po, emb1, Lacc + 0);          // loss1
    proj(emb1, 46, po); sce(po, emb2, Lacc + 4);          // loss5
    add2(seb[0], seb[2], tgt); proj(seb[1], 52, po); sce(po, tgt, Lacc + 1);  // loss2
    add2(seb[1], seb[2], tgt); proj(seb[0], 58, po); sce(po, tgt, Lacc + 2);  // loss3
    add2(seb[0], seb[1], tgt); proj(seb[2], 64, po); sce(po, tgt, Lacc + 3);  // loss4

    hg_combine<<<1, 1, 0, stream>>>(Lacc, (float*)d_out);
}
